// GlobalFeatureGAT_23338852286626
// MI455X (gfx1250) — matrix operand order repaired and checked by reading, not compiled, not run on hardware
//
#include <hip/hip_runtime.h>
#include <hip/hip_bf16.h>

typedef __attribute__((ext_vector_type(16))) _Float16 v16h;
typedef __attribute__((ext_vector_type(8)))  _Float16 v8h;
typedef __attribute__((ext_vector_type(8)))  float    v8f;

#define NN 20000           // nodes
#define NE 320000          // raw edges
#define ET (NE + NN)       // edges incl. self loops
#define SEQL 20
#define EMBD 32
#define HIDC 128
#define INDIM 176
#define KPAD1 192
#define NC1 512            // 4 heads * 128
#define KS1 (KPAD1 / 32)   // 6
#define KPAD2 512
#define NC2 128
#define KS2 (KPAD2 / 32)   // 16
#define NG 64

// ---------- ordered-uint encoding for float atomicMax ----------
__device__ __forceinline__ unsigned enc_ord(float f) {
  unsigned u = __float_as_uint(f);
  return (u & 0x80000000u) ? ~u : (u | 0x80000000u);
}
__device__ __forceinline__ float dec_ord(unsigned u) {
  unsigned b = (u & 0x80000000u) ? (u ^ 0x80000000u) : ~u;
  return __uint_as_float(b);
}

// ---------- generic fill ----------
__global__ void fill_u32_kernel(unsigned* __restrict__ p, unsigned v, int n) {
  int i = blockIdx.x * blockDim.x + threadIdx.x;
  if (i < n) p[i] = v;
}

// ---------- pack f32 weights [Kact, Ncols] -> WMMA-B f16 layout ----------
// Bp[((ntile*ksteps + kk)*32 + lane)*16 + i] = W[kk*32 + (lane>>4)*8 + (i&7) + (i>>3)*16][ntile*16 + (lane&15)]
__global__ void pack_b_kernel(const float* __restrict__ W, _Float16* __restrict__ Bp,
                              int Kact, int Ncols, int ksteps) {
  int gid = blockIdx.x * blockDim.x + threadIdx.x;
  int total = (Ncols >> 4) * ksteps * 512;
  if (gid >= total) return;
  int i     = gid & 15;
  int lane  = (gid >> 4) & 31;
  int kk    = (gid >> 9) % ksteps;
  int ntile = (gid >> 9) / ksteps;
  int hi = lane >> 4;
  int k  = kk * 32 + hi * 8 + (i & 7) + (i >> 3) * 16;
  int nn = ntile * 16 + (lane & 15);
  float v = (k < Kact) ? W[(size_t)k * Ncols + nn] : 0.0f;
  Bp[gid] = (_Float16)v;
}

// ---------- fused embedding masked-mean + concat + LayerNorm -> f16 [NN, 192] ----------
__global__ void feats_ln_kernel(const float* __restrict__ xs,
                                const int* __restrict__ xop, const int* __restrict__ xsrc,
                                const int* __restrict__ xsink, const int* __restrict__ xstr,
                                const int* __restrict__ xpay,
                                const float* __restrict__ eop, const float* __restrict__ esrc,
                                const float* __restrict__ esink, const float* __restrict__ estr,
                                const float* __restrict__ epay,
                                const float* __restrict__ gamma, const float* __restrict__ beta,
                                _Float16* __restrict__ hA) {
  __shared__ float sfe[8][INDIM];
  int lane = threadIdx.x & 31;
  int wave = threadIdx.x >> 5;
  int n = blockIdx.x * 8 + wave;
  if (n >= NN) return;
  float* f = sfe[wave];
  if (lane < 16) f[lane] = xs[n * 16 + lane];
  const int*   idxs[5] = {xop, xsrc, xsink, xstr, xpay};
  const float* tabs[5] = {eop, esrc, esink, estr, epay};
  #pragma unroll
  for (int t = 0; t < 5; ++t) {
    float acc = 0.f, cnt = 0.f;
    const int* ip = idxs[t] + (size_t)n * SEQL;
    const float* tb = tabs[t];
    for (int l = 0; l < SEQL; ++l) {
      int id = ip[l];
      if (id != 0) { acc += tb[(size_t)id * EMBD + lane]; cnt += 1.f; }
    }
    f[16 + t * EMBD + lane] = acc / (cnt + 1e-9f);
  }
  __syncthreads();
  // wave-level LayerNorm over 176 features
  float s = 0.f;
  for (int c = lane; c < INDIM; c += 32) s += f[c];
  #pragma unroll
  for (int off = 16; off > 0; off >>= 1) s += __shfl_xor(s, off, 32);
  float mu = s * (1.0f / INDIM);
  float v = 0.f;
  for (int c = lane; c < INDIM; c += 32) { float d = f[c] - mu; v += d * d; }
  #pragma unroll
  for (int off = 16; off > 0; off >>= 1) v += __shfl_xor(v, off, 32);
  float rstd = rsqrtf(v * (1.0f / INDIM) + 1e-5f);
  _Float16* hrow = hA + (size_t)n * KPAD1;
  for (int c = lane; c < INDIM; c += 32)
    hrow[c] = (_Float16)(((f[c] - mu) * rstd) * gamma[c] + beta[c]);
  for (int c = INDIM + lane; c < KPAD1; c += 32)
    hrow[c] = (_Float16)0.f;
}

__device__ __forceinline__ v16h load_a_frag(const _Float16* __restrict__ p) {
  v8h lo = *(const v8h*)(p);
  v8h hi = *(const v8h*)(p + 16);
  return __builtin_shufflevector(lo, hi, 0,1,2,3,4,5,6,7,8,9,10,11,12,13,14,15);
}

// ---------- WMMA GEMM: C[M, Ncols] = A_f16[M, Kpad] @ Bpacked ----------
// one wave -> 16 rows x (JT*16) cols; fully unrolled, software-pipelined:
// k+1's A/B loads issue before k's WMMAs. __launch_bounds__(THREADS,1) lifts
// the VGPR cap so the double-buffer registers are not coalesced away.
template <int KSTEPS, int JT, int THREADS>
__global__ __launch_bounds__(THREADS, 1)
void gemm_wmma_kernel(const _Float16* __restrict__ A,
                      const _Float16* __restrict__ Bp,
                      float* __restrict__ C,
                      int Kpad, int Ncols) {
  int lane  = threadIdx.x & 31;
  int wave  = threadIdx.x >> 5;
  int mtile = blockIdx.x;
  int hi = lane >> 4;
  int m  = mtile * 16 + (lane & 15);
  int ntile0 = wave * JT;

  // A layout (16-bit 16x32): lanes 0-15 K=0..7 & 16..23, lanes 16-31 K=8..15 & 24..31
  const _Float16* abase = A + (size_t)m * Kpad + hi * 8;
  // speculative prefetch of next M-tile's A rows (dropped if out of range)
  __builtin_prefetch(abase + (size_t)16 * Kpad, 0, 0);
  const _Float16* bbase = Bp + ((size_t)ntile0 * KSTEPS * 32 + lane) * 16;

  v8f  acc[JT];
  v16h bcur[JT], bnxt[JT];
  #pragma unroll
  for (int j = 0; j < JT; ++j) {
    acc[j]  = (v8f){0.f, 0.f, 0.f, 0.f, 0.f, 0.f, 0.f, 0.f};
    bcur[j] = *(const v16h*)(bbase + (j * KSTEPS) * 512);   // kk = 0
  }
  v16h acur = load_a_frag(abase);
  v16h anxt;

  #pragma unroll
  for (int kk = 0; kk < KSTEPS; ++kk) {
    if (kk + 1 < KSTEPS) {
      anxt = load_a_frag(abase + (kk + 1) * 32);
      #pragma unroll
      for (int j = 0; j < JT; ++j)
        bnxt[j] = *(const v16h*)(bbase + (j * KSTEPS + kk + 1) * 512);
    }
    #pragma unroll
    for (int j = 0; j < JT; ++j)
      acc[j] = __builtin_amdgcn_wmma_f32_16x16x32_f16(
          /*neg_a=*/false, acur, /*neg_b=*/false, bcur[j],
          /*c_mod=*/(short)0, acc[j], /*reuse_a=*/false, /*reuse_b=*/false);
    if (kk + 1 < KSTEPS) {
      acur = anxt;
      #pragma unroll
      for (int j = 0; j < JT; ++j) bcur[j] = bnxt[j];
    }
  }

  // C/D layout: VGPR r -> M = r + 8*(lane>=16); N = lane&15
  int mrow = mtile * 16 + 8 * hi;
  #pragma unroll
  for (int j = 0; j < JT; ++j) {
    int n0 = (ntile0 + j) * 16 + (lane & 15);
    #pragma unroll
    for (int r = 0; r < 8; ++r)
      C[(size_t)(mrow + r) * Ncols + n0] = acc[j][r];
  }
}

// ---------- per-node attention scalars: a = sum_c xw[n,h,c]*att[h,c] ----------
__global__ void att_nodes_kernel(const float* __restrict__ xw,
                                 const float* __restrict__ att_s, const float* __restrict__ att_d,
                                 float* __restrict__ as_, float* __restrict__ ad_, int H) {
  int lane = threadIdx.x & 31;
  int wave = threadIdx.x >> 5;
  int n = blockIdx.x * (blockDim.x >> 5) + wave;
  if (n >= NN) return;
  for (int h = 0; h < H; ++h) {
    const float* row = xw + ((size_t)n * H + h) * HIDC;
    float ss = 0.f, sd = 0.f;
    #pragma unroll
    for (int j = 0; j < 4; ++j) {
      int c = lane + j * 32;
      float v = row[c];
      ss += v * att_s[h * HIDC + c];
      sd += v * att_d[h * HIDC + c];
    }
    #pragma unroll
    for (int off = 16; off > 0; off >>= 1) {
      ss += __shfl_xor(ss, off, 32);
      sd += __shfl_xor(sd, off, 32);
    }
    if (lane == 0) { as_[n * H + h] = ss; ad_[n * H + h] = sd; }
  }
}

__device__ __forceinline__ void edge_sd(const int* __restrict__ ei, int i, int& s, int& d) {
  if (i < NE) { s = ei[i]; d = ei[NE + i]; }
  else        { s = i - NE; d = s; }           // self loop
}

// ---------- edge pass 1: leaky-relu logits + segment max ----------
__global__ void edge_logits_kernel(const int* __restrict__ ei,
                                   const float* __restrict__ as_, const float* __restrict__ ad_,
                                   float* __restrict__ e, unsigned* __restrict__ nmax, int H) {
  int gid = blockIdx.x * blockDim.x + threadIdx.x;
  if (gid >= ET * H) return;
  int h = gid % H, i = gid / H;
  int s, d; edge_sd(ei, i, s, d);
  float val = as_[s * H + h] + ad_[d * H + h];
  val = (val > 0.f) ? val : 0.2f * val;        // leaky_relu(0.2)
  e[gid] = val;
  atomicMax(&nmax[d * H + h], enc_ord(val));
}

// ---------- edge pass 2: exp(e - max) + segment sum ----------
__global__ void edge_exp_kernel(const int* __restrict__ ei,
                                float* __restrict__ e, const unsigned* __restrict__ nmax,
                                float* __restrict__ denom, int H) {
  int gid = blockIdx.x * blockDim.x + threadIdx.x;
  if (gid >= ET * H) return;
  int h = gid % H, i = gid / H;
  int s, d; edge_sd(ei, i, s, d);
  float ex = __expf(e[gid] - dec_ord(nmax[d * H + h]));
  e[gid] = ex;
  atomicAdd(&denom[d * H + h], ex);
}

// ---------- edge pass 3: scatter alpha-weighted messages (wave per edge-head) ----------
__global__ void edge_scatter_kernel(const int* __restrict__ ei,
                                    const float* __restrict__ e, const float* __restrict__ denom,
                                    const float* __restrict__ xw, float* __restrict__ out, int H) {
  int lane = threadIdx.x & 31;
  int wid = (blockIdx.x * blockDim.x + threadIdx.x) >> 5;
  if (wid >= ET * H) return;
  int h = wid % H, i = wid / H;
  int s, d; edge_sd(ei, i, s, d);
  float alpha = e[wid] / (denom[d * H + h] + 1e-16f);
  const float* srow = xw  + ((size_t)s * H + h) * HIDC;
  float*       drow = out + ((size_t)d * H + h) * HIDC;
  #pragma unroll
  for (int j = 0; j < 4; ++j) {
    int c = lane + j * 32;
    atomicAdd(&drow[c], srow[c] * alpha);
  }
}

// ---------- bias + ELU -> f16 (feeds next GEMM) ----------
__global__ void bias_elu_h_kernel(const float* __restrict__ agg, const float* __restrict__ bias,
                                  _Float16* __restrict__ xh, int cols) {
  int gid = blockIdx.x * blockDim.x + threadIdx.x;
  if (gid >= NN * cols) return;
  int c = gid % cols;
  float v = agg[gid] + bias[c];
  v = (v > 0.f) ? v : (__expf(v) - 1.f);
  xh[gid] = (_Float16)v;
}

// ---------- bias + ELU + graph segment-max pool ----------
__global__ void bias_elu_pool_kernel(const float* __restrict__ agg, const float* __restrict__ bias,
                                     const int* __restrict__ batch, unsigned* __restrict__ pooled) {
  int gid = blockIdx.x * blockDim.x + threadIdx.x;
  if (gid >= NN * HIDC) return;
  int n = gid / HIDC, c = gid % HIDC;
  float v = agg[gid] + bias[c];
  v = (v > 0.f) ? v : (__expf(v) - 1.f);
  atomicMax(&pooled[batch[n] * HIDC + c], enc_ord(v));
}

// ---------- classifier: relu(pooled@W1+b1)@W2+b2 -> [64,2] ----------
__global__ void classifier_kernel(const unsigned* __restrict__ pooled,
                                  const float* __restrict__ w1, const float* __restrict__ b1,
                                  const float* __restrict__ w2, const float* __restrict__ b2,
                                  float* __restrict__ outp) {
  __shared__ float hc[NG * 64];
  int tid = threadIdx.x;               // blockDim.x == 128
  if (tid < 64) {
    int j = tid;
    for (int g = 0; g < NG; ++g) {
      float acc = b1[j];
      for (int c = 0; c < HIDC; ++c)
        acc += dec_ord(pooled[g * HIDC + c]) * w1[c * 64 + j];
      hc[g * 64 + j] = (acc > 0.f) ? acc : 0.f;
    }
  }
  __syncthreads();
  int g = tid >> 1, k = tid & 1;
  float acc = b2[k];
  for (int j = 0; j < 64; ++j) acc += hc[g * 64 + j] * w2[j * 2 + k];
  outp[g * 2 + k] = acc;
}

extern "C" void kernel_launch(void* const* d_in, const int* in_sizes, int n_in,
                              void* d_out, int out_size, void* d_ws, size_t ws_size,
                              hipStream_t stream) {
  (void)in_sizes; (void)n_in; (void)out_size; (void)ws_size;
  const float* x_scalar = (const float*)d_in[0];
  const int*   x_opcode = (const int*)d_in[1];
  const int*   x_source = (const int*)d_in[2];
  const int*   x_sink   = (const int*)d_in[3];
  const int*   x_string = (const int*)d_in[4];
  const int*   x_payload= (const int*)d_in[5];
  const int*   edge_idx = (const int*)d_in[6];
  const int*   batch    = (const int*)d_in[7];
  const float* emb_op   = (const float*)d_in[8];
  const float* emb_src  = (const float*)d_in[9];
  const float* emb_sink = (const float*)d_in[10];
  const float* emb_str  = (const float*)d_in[11];
  const float* emb_pay  = (const float*)d_in[12];
  const float* ln_gamma = (const float*)d_in[13];
  const float* ln_beta  = (const float*)d_in[14];
  const float* W1       = (const float*)d_in[15];
  const float* att_src1 = (const float*)d_in[16];
  const float* att_dst1 = (const float*)d_in[17];
  const float* b1       = (const float*)d_in[18];
  const float* W2       = (const float*)d_in[19];
  const float* att_src2 = (const float*)d_in[20];
  const float* att_dst2 = (const float*)d_in[21];
  const float* b2       = (const float*)d_in[22];
  const float* cls_w1   = (const float*)d_in[23];
  const float* cls_b1   = (const float*)d_in[24];
  const float* cls_w2   = (const float*)d_in[25];
  const float* cls_b2   = (const float*)d_in[26];
  float* out = (float*)d_out;

  // ---- carve workspace ----
  char* ws = (char*)d_ws;
  size_t off = 0;
  auto carve = [&](size_t bytes) -> void* {
    void* p = ws + off;
    off += (bytes + 255) & ~(size_t)255;
    return p;
  };
  _Float16* hA    = (_Float16*)carve((size_t)NN * KPAD1 * 2);        // LN'd feats, f16
  _Float16* W1p   = (_Float16*)carve((size_t)(NC1/16) * KS1 * 512 * 2);
  _Float16* W2p   = (_Float16*)carve((size_t)(NC2/16) * KS2 * 512 * 2);
  float*    xw1   = (float*)carve((size_t)NN * NC1 * 4);
  float*    as1   = (float*)carve((size_t)NN * 4 * 4);
  float*    ad1   = (float*)carve((size_t)NN * 4 * 4);
  float*    e1    = (float*)carve((size_t)ET * 4 * 4);
  unsigned* nmax1 = (unsigned*)carve((size_t)NN * 4 * 4);
  float*    den1  = (float*)carve((size_t)NN * 4 * 4);
  float*    agg1  = (float*)carve((size_t)NN * NC1 * 4);
  _Float16* x1h   = (_Float16*)carve((size_t)NN * NC1 * 2);
  float*    xw2   = (float*)carve((size_t)NN * NC2 * 4);
  float*    as2   = (float*)carve((size_t)NN * 4);
  float*    ad2   = (float*)carve((size_t)NN * 4);
  float*    e2    = (float*)carve((size_t)ET * 4);
  unsigned* nmax2 = (unsigned*)carve((size_t)NN * 4);
  float*    den2  = (float*)carve((size_t)NN * 4);
  float*    agg2  = (float*)carve((size_t)NN * NC2 * 4);
  unsigned* pooled= (unsigned*)carve((size_t)NG * HIDC * 4);

  auto fill0 = [&](void* p, size_t n_u32) {
    int n = (int)n_u32;
    fill_u32_kernel<<<(n + 255) / 256, 256, 0, stream>>>((unsigned*)p, 0u, n);
  };

  // ---- weight packing (deterministic, done every call) ----
  {
    int t1 = (NC1 / 16) * KS1 * 512;
    pack_b_kernel<<<(t1 + 255) / 256, 256, 0, stream>>>(W1, W1p, INDIM, NC1, KS1);
    int t2 = (NC2 / 16) * KS2 * 512;
    pack_b_kernel<<<(t2 + 255) / 256, 256, 0, stream>>>(W2, W2p, KPAD2, NC2, KS2);
  }

  // ---- zero accumulators (0 bits == float 0 == ordered-min) ----
  fill0(nmax1, (size_t)NN * 4);  fill0(den1, (size_t)NN * 4);  fill0(agg1, (size_t)NN * NC1);
  fill0(nmax2, (size_t)NN);      fill0(den2, (size_t)NN);      fill0(agg2, (size_t)NN * NC2);
  fill0(pooled, (size_t)NG * HIDC);

  // ---- features + LayerNorm ----
  feats_ln_kernel<<<NN / 8, 256, 0, stream>>>(
      x_scalar, x_opcode, x_source, x_sink, x_string, x_payload,
      emb_op, emb_src, emb_sink, emb_str, emb_pay, ln_gamma, ln_beta, hA);

  // ---- GAT layer 1 ----
  // 8 waves/block, each wave 16x64 of the 16x512 strip (double-buffered)
  gemm_wmma_kernel<KS1, 4, 256><<<NN / 16, 256, 0, stream>>>(hA, W1p, xw1, KPAD1, NC1);
  att_nodes_kernel<<<(NN + 7) / 8, 256, 0, stream>>>(xw1, att_src1, att_dst1, as1, ad1, 4);
  {
    int n = ET * 4;
    edge_logits_kernel<<<(n + 255) / 256, 256, 0, stream>>>(edge_idx, as1, ad1, e1, nmax1, 4);
    edge_exp_kernel<<<(n + 255) / 256, 256, 0, stream>>>(edge_idx, e1, nmax1, den1, 4);
    edge_scatter_kernel<<<(n + 7) / 8, 256, 0, stream>>>(edge_idx, e1, den1, xw1, agg1, 4);
  }
  {
    int n = NN * NC1;
    bias_elu_h_kernel<<<(n + 255) / 256, 256, 0, stream>>>(agg1, b1, x1h, NC1);
  }

  // ---- GAT layer 2 ----
  // 2 waves/block cover the 16x128 strip (double-buffered)
  gemm_wmma_kernel<KS2, 4, 64><<<NN / 16, 64, 0, stream>>>(x1h, W2p, xw2, KPAD2, NC2);
  att_nodes_kernel<<<(NN + 7) / 8, 256, 0, stream>>>(xw2, att_src2, att_dst2, as2, ad2, 1);
  {
    int n = ET;
    edge_logits_kernel<<<(n + 255) / 256, 256, 0, stream>>>(edge_idx, as2, ad2, e2, nmax2, 1);
    edge_exp_kernel<<<(n + 255) / 256, 256, 0, stream>>>(edge_idx, e2, nmax2, den2, 1);
    edge_scatter_kernel<<<(n + 7) / 8, 256, 0, stream>>>(edge_idx, e2, den2, xw2, agg2, 1);
  }
  {
    int n = NN * NC2;
    bias_elu_pool_kernel<<<(n + 255) / 256, 256, 0, stream>>>(agg2, b2, batch, pooled);
  }

  // ---- classifier ----
  classifier_kernel<<<1, 128, 0, stream>>>(pooled, cls_w1, cls_b1, cls_w2, cls_b2, out);
}
